// SelfAttentionHead_80427557585251
// MI455X (gfx1250) — matrix operand order repaired and checked by reading, not compiled, not run on hardware
//
#include <hip/hip_runtime.h>

// ---------------------------------------------------------------------------
// Single-head attention, B=8 T=2048 C=2048 H=128, bf16 WMMA path (gfx1250).
// Kernel 0a: x fp32 -> bf16 (once)        Kernel 0b: W -> bf16, transposed
// Kernel 1 : fused QKV projection         Kernel 2 : flash attention
// Tile staging: TDM (tensor_load_to_lds, TENSORcnt, hardware row padding)
// when available; else per-lane GLOBAL_LOAD_ASYNC_TO_LDS_B128 (ASYNCcnt);
// else plain LDS copies.  Double-buffered so DMA overlaps WMMA.
// LDS row strides padded (56/136/72 ushorts) for conflict-free b128 gathers.
// ---------------------------------------------------------------------------

#define BATCH 8
#define SEQ   2048
#define CDIM  2048
#define HDIM  128
#define QSCALE 0.08838834764831845f   // 1/sqrt(128)

#define XS_STR 56    // QKV tile row stride (ushorts): 64B data + 48B pad
#define KS_STR 136   // K tile row stride: 256B data + 16B pad
#define VS_STR 72    // V^T / Pbuf row stride: 128B data + 16B pad

typedef __attribute__((ext_vector_type(16))) __bf16 v16bf;
typedef __attribute__((ext_vector_type(8)))  float  v8f;

struct BF16Frag { uint4 lo, hi; };

static __device__ __forceinline__ v16bf make_frag(uint4 lo, uint4 hi) {
    BF16Frag s{lo, hi};
    return __builtin_bit_cast(v16bf, s);
}

static __device__ __forceinline__ unsigned short f2bf(float f) {
    return __builtin_bit_cast(unsigned short, static_cast<__bf16>(f));
}

static __device__ __forceinline__ v8f wmma_bf16(v16bf a, v16bf b, v8f c) {
    return __builtin_amdgcn_wmma_f32_16x16x32_bf16(
        false, a, false, b, (short)0, c, false, false);
}

// ----------------------- staging path selection -----------------------------
#if defined(__AMDGCN__) && __has_builtin(__builtin_amdgcn_tensor_load_to_lds) && __has_builtin(__builtin_amdgcn_s_wait_tensorcnt)
#define USE_TDM 1
#else
#define USE_TDM 0
#endif

#if defined(__AMDGCN__) && __has_builtin(__builtin_amdgcn_global_load_async_to_lds_b128) && __has_builtin(__builtin_amdgcn_s_wait_asynccnt)
#define USE_ASYNC_LDS 1
#else
#define USE_ASYNC_LDS 0
#endif

#if USE_TDM
typedef unsigned tdm_v4u __attribute__((vector_size(16)));
typedef int      tdm_v8i __attribute__((vector_size(32)));
typedef int      tdm_v4i __attribute__((vector_size(16)));

// D# group-1 word 0: data_size=2B (bits17:16=1), pad_enable (bit20),
// pad_interval (bits24:22), pad_amount (bits31:25).
#define TDM_W0_K  0x07510000u   // 256B rows, +16B pad  (interval 5, amount 3)
#define TDM_W0_V  0x07110000u   // 128B rows, +16B pad  (interval 4, amount 3)
#define TDM_W0_X  0x16D10000u   //  64B rows, +48B pad  (interval 3, amount 11)

// 2-D tile load: tile_d0 elements per row (dim-0), tile_d1 rows, row stride
// stride0 (elements).  Descriptor layout per CDNA5 ISA.
static __device__ __forceinline__ void tdm_load_2d(
    unsigned lds_addr, const void* gaddr,
    unsigned tensor_d0, unsigned tensor_d1,
    unsigned tile_d0, unsigned tile_d1,
    unsigned stride0, unsigned w0) {
    unsigned long long ga = (unsigned long long)(size_t)gaddr;
    tdm_v4u g0 = { 1u,                                    // count=1, user mode
                   lds_addr,                              // LDS byte address
                   (unsigned)ga,                          // global_addr[31:0]
                   (unsigned)((ga >> 32) & 0x1FFFFFFu) | 0x80000000u }; // +type=2
    tdm_v8i g1;
    g1[0] = (int)w0;                                      // mask=0, size, pad
    g1[1] = (int)((tensor_d0 & 0xFFFFu) << 16);           // barrier=0, dim0 lo
    g1[2] = (int)((tensor_d0 >> 16) | ((tensor_d1 & 0xFFFFu) << 16));
    g1[3] = (int)((tensor_d1 >> 16) | (tile_d0 << 16));
    g1[4] = (int)tile_d1;                                 // tile_d2 = 0
    g1[5] = (int)stride0;                                 // stride0[31:0]
    g1[6] = 0;                                            // stride0 hi, stride1 lo
    g1[7] = 0;
    tdm_v4i z4 = { 0, 0, 0, 0 };
    tdm_v8i z8 = { 0, 0, 0, 0, 0, 0, 0, 0 };
    __builtin_amdgcn_tensor_load_to_lds(g0, g1, z4, z4, z8, 0);
}

template <int N>
static __device__ __forceinline__ void tdm_wait_n() {
    __builtin_amdgcn_s_wait_tensorcnt(N);
}
#endif

#if USE_ASYNC_LDS
typedef int v4i_ __attribute__((vector_size(16)));
typedef __attribute__((address_space(1))) v4i_* async_gptr;
typedef __attribute__((address_space(3))) v4i_* async_lptr;
#endif

static __device__ __forceinline__ void stage_b128(unsigned short* lds,
                                                  const unsigned short* g) {
#if USE_ASYNC_LDS
    __builtin_amdgcn_global_load_async_to_lds_b128(
        (async_gptr)(unsigned long long)(size_t)g,
        (async_lptr)(unsigned)(size_t)lds,
        0, 0);
#else
    *(uint4*)lds = *(const uint4*)g;
#endif
}

template <int N>
static __device__ __forceinline__ void stage_wait_n() {
#if USE_ASYNC_LDS
    __builtin_amdgcn_s_wait_asynccnt(N);
#endif
}

// ---------------------------------------------------------------------------
// Kernel 0a: x fp32 -> bf16.  Each thread: 8 floats -> one uint4 store.
// ---------------------------------------------------------------------------
__global__ void __launch_bounds__(256)
cvt_x_kernel(const float* __restrict__ x, unsigned short* __restrict__ xbf) {
    size_t idx = (size_t)blockIdx.x * 256 + threadIdx.x;
    const float4* x4 = (const float4*)x;
    float4 a = x4[idx * 2 + 0];
    float4 b = x4[idx * 2 + 1];
    unsigned short o[8] = { f2bf(a.x), f2bf(a.y), f2bf(a.z), f2bf(a.w),
                            f2bf(b.x), f2bf(b.y), f2bf(b.z), f2bf(b.w) };
    *(uint4*)(xbf + idx * 8) = *(const uint4*)o;
}

// ---------------------------------------------------------------------------
// Kernel 0b: W [C][H] fp32 -> WT [3][H][C] bf16 (transposed).
// ---------------------------------------------------------------------------
__global__ void __launch_bounds__(256)
cvt_w_kernel(const float* __restrict__ Wq, const float* __restrict__ Wk,
             const float* __restrict__ Wv, unsigned short* __restrict__ WT) {
    const int p = blockIdx.y;
    const float* W = (p == 0) ? Wq : (p == 1) ? Wk : Wv;
    int idx = blockIdx.x * 256 + threadIdx.x;
    int n = idx >> 11;
    int c = idx & (CDIM - 1);
    WT[((size_t)p * HDIM + n) * CDIM + c] = f2bf(W[(size_t)c * HDIM + n]);
}

// ---------------------------------------------------------------------------
// Kernel 1: QKV projection.  grid = (M/128, 3), block = 256 (8 waves).
// ---------------------------------------------------------------------------
__global__ void __launch_bounds__(256)
qkv_proj_kernel(const unsigned short* __restrict__ xbf,
                const unsigned short* __restrict__ WT,
                unsigned short* __restrict__ qws,
                unsigned short* __restrict__ kws,
                unsigned short* __restrict__ vT) {
    __shared__ alignas(16) unsigned short xs[2][128][XS_STR];
    __shared__ alignas(16) unsigned short wsT[2][128][XS_STR];

    const int tid  = threadIdx.x;
    const int wave = tid >> 5;
    const int lane = tid & 31;
    const int rowBase = blockIdx.x * 128;
    const int proj = blockIdx.y;
    const unsigned short* xrow = xbf + (size_t)rowBase * CDIM;
    const unsigned short* wrow = WT + (size_t)proj * HDIM * CDIM;

    v8f acc[8];
#pragma unroll
    for (int t = 0; t < 8; ++t) acc[t] = (v8f)0.0f;

#if !USE_TDM
    const int sr  = (tid + 0) >> 2;
    const int sr2 = (tid + 256) >> 2;
    const int sc  = (tid & 3) << 3;
#endif

    // Prologue: stage chunk 0 into buffer 0.
#if USE_TDM
    if (wave == 0) {
        tdm_load_2d((unsigned)(size_t)&xs[0][0][0],  xrow, CDIM, 16384, 32, 128, CDIM, TDM_W0_X);
        tdm_load_2d((unsigned)(size_t)&wsT[0][0][0], wrow, CDIM, 16384, 32, 128, CDIM, TDM_W0_X);
    }
#else
    stage_b128(&xs[0][sr][sc],   xrow + (size_t)sr * CDIM + sc);
    stage_b128(&xs[0][sr2][sc],  xrow + (size_t)sr2 * CDIM + sc);
    stage_b128(&wsT[0][sr][sc],  wrow + (size_t)sr * CDIM + sc);
    stage_b128(&wsT[0][sr2][sc], wrow + (size_t)sr2 * CDIM + sc);
#endif

    for (int kc = 0; kc < CDIM; kc += 32) {
        const int p = (kc >> 5) & 1;
        const int kn = kc + 32;
#if USE_TDM
        if (wave == 0) {
            if (kn < CDIM) {
                tdm_load_2d((unsigned)(size_t)&xs[1 - p][0][0],  xrow + kn,
                            CDIM, 16384, 32, 128, CDIM, TDM_W0_X);
                tdm_load_2d((unsigned)(size_t)&wsT[1 - p][0][0], wrow + kn,
                            CDIM, 16384, 32, 128, CDIM, TDM_W0_X);
                tdm_wait_n<2>();   // retire current buffer's two tiles
            } else {
                tdm_wait_n<0>();
            }
        }
#else
        if (kn < CDIM) {
            stage_b128(&xs[1 - p][sr][sc],   xrow + (size_t)sr * CDIM + kn + sc);
            stage_b128(&xs[1 - p][sr2][sc],  xrow + (size_t)sr2 * CDIM + kn + sc);
            stage_b128(&wsT[1 - p][sr][sc],  wrow + (size_t)sr * CDIM + kn + sc);
            stage_b128(&wsT[1 - p][sr2][sc], wrow + (size_t)sr2 * CDIM + kn + sc);
            stage_wait_n<4>();
        } else {
            stage_wait_n<0>();
        }
#endif
        __syncthreads();

        // A fragment (16x32): lane<16 holds K {0-7,16-23}, lane>=16 {8-15,24-31}.
        const int arow = wave * 16 + (lane & 15);
        const int asel = (lane >> 4) * 8;
        uint4 alo = *(const uint4*)&xs[p][arow][asel];
        uint4 ahi = *(const uint4*)&xs[p][arow][asel + 16];
        v16bf afrag = make_frag(alo, ahi);

        // 8 B tiles (32x16): lane<16 holds K {0-7,16-23}, lane>=16 {8-15,24-31}.
        const int bsel = (lane >> 4) * 8;
#pragma unroll
        for (int t = 0; t < 8; ++t) {
            int n = t * 16 + (lane & 15);
            uint4 blo = *(const uint4*)&wsT[p][n][bsel];
            uint4 bhi = *(const uint4*)&wsT[p][n][bsel + 16];
            acc[t] = wmma_bf16(afrag, make_frag(blo, bhi), acc[t]);
        }
        __syncthreads();   // all reads of buffer p done before refill
    }

    const float oscale = (proj == 0) ? QSCALE : 1.0f;
    if (proj == 2) {
#pragma unroll
        for (int t = 0; t < 8; ++t) {
            int col = t * 16 + (lane & 15);
#pragma unroll
            for (int i = 0; i < 8; ++i) {
                int row = rowBase + wave * 16 + i + 8 * (lane >> 4);
                int bb = row >> 11, tt = row & (SEQ - 1);
                vT[((size_t)bb * HDIM + col) * SEQ + tt] = f2bf(acc[t][i]);
            }
        }
    } else {
        unsigned short* outb = (proj == 0) ? qws : kws;
#pragma unroll
        for (int t = 0; t < 8; ++t) {
            int col = t * 16 + (lane & 15);
#pragma unroll
            for (int i = 0; i < 8; ++i) {
                int row = rowBase + wave * 16 + i + 8 * (lane >> 4);
                outb[(size_t)row * HDIM + col] = f2bf(acc[t][i] * oscale);
            }
        }
    }
}

// ---------------------------------------------------------------------------
// Kernel 2: flash attention.  grid = (T/128, B), block = 256 (8 waves).
// ---------------------------------------------------------------------------
__global__ void __launch_bounds__(256)
flash_attn_kernel(const unsigned short* __restrict__ qws,
                  const unsigned short* __restrict__ kws,
                  const unsigned short* __restrict__ vT,
                  float* __restrict__ out) {
    __shared__ alignas(16) unsigned short Ks[2][64][KS_STR];
    __shared__ alignas(16) unsigned short VsT[2][128][VS_STR];
    __shared__ alignas(16) unsigned short Pbuf[8][16][VS_STR];

    const int tid  = threadIdx.x;
    const int wave = tid >> 5;
    const int lane = tid & 31;
    const int b = blockIdx.y;
    const int qBase = blockIdx.x * 128 + wave * 16;
    const size_t bOff = (size_t)b * SEQ * HDIM;
    const size_t vOff = (size_t)b * HDIM * SEQ;

    // Q fragments (pre-scaled by 1/sqrt(H)).
    v16bf qfrag[4];
    {
        const int row = qBase + (lane & 15);
        const int sel = (lane >> 4) * 8;
        const unsigned short* qrow = qws + bOff + (size_t)row * HDIM;
#pragma unroll
        for (int c = 0; c < 4; ++c) {
            uint4 lo = *(const uint4*)(qrow + c * 32 + sel);
            uint4 hi = *(const uint4*)(qrow + c * 32 + sel + 16);
            qfrag[c] = make_frag(lo, hi);
        }
    }

    v8f oacc[8];
#pragma unroll
    for (int t = 0; t < 8; ++t) oacc[t] = (v8f)0.0f;
    float mrow[8], lrow[8];
#pragma unroll
    for (int i = 0; i < 8; ++i) { mrow[i] = -__builtin_inff(); lrow[i] = 0.0f; }

#if !USE_TDM
    const int kr[4] = { (tid + 0) >> 4, (tid + 256) >> 4, (tid + 512) >> 4, (tid + 768) >> 4 };
    const int kc16 = (tid & 15) << 3;
    const int vh[4] = { (tid + 0) >> 3, (tid + 256) >> 3, (tid + 512) >> 3, (tid + 768) >> 3 };
    const int vc8 = (tid & 7) << 3;
#endif

    // Prologue: stage key block 0 into buffer 0.
#if USE_TDM
    if (wave == 0) {
        tdm_load_2d((unsigned)(size_t)&Ks[0][0][0], kws + bOff,
                    HDIM, SEQ, 128, 64, HDIM, TDM_W0_K);
        tdm_load_2d((unsigned)(size_t)&VsT[0][0][0], vT + vOff,
                    SEQ, HDIM, 64, 128, SEQ, TDM_W0_V);
    }
#else
#pragma unroll
    for (int i = 0; i < 4; ++i)
        stage_b128(&Ks[0][kr[i]][kc16], kws + bOff + (size_t)kr[i] * HDIM + kc16);
#pragma unroll
    for (int i = 0; i < 4; ++i)
        stage_b128(&VsT[0][vh[i]][vc8], vT + vOff + (size_t)vh[i] * SEQ + vc8);
#endif

    for (int kb = 0; kb < SEQ; kb += 64) {
        const int p = (kb >> 6) & 1;
        const int kn = kb + 64;
#if USE_TDM
        if (wave == 0) {
            if (kn < SEQ) {
                tdm_load_2d((unsigned)(size_t)&Ks[1 - p][0][0],
                            kws + bOff + (size_t)kn * HDIM,
                            HDIM, SEQ, 128, 64, HDIM, TDM_W0_K);
                tdm_load_2d((unsigned)(size_t)&VsT[1 - p][0][0],
                            vT + vOff + kn,
                            SEQ, HDIM, 64, 128, SEQ, TDM_W0_V);
                tdm_wait_n<2>();   // retire current buffer's two tiles
            } else {
                tdm_wait_n<0>();
            }
        }
#else
        if (kn < SEQ) {
#pragma unroll
            for (int i = 0; i < 4; ++i)
                stage_b128(&Ks[1 - p][kr[i]][kc16],
                           kws + bOff + (size_t)(kn + kr[i]) * HDIM + kc16);
#pragma unroll
            for (int i = 0; i < 4; ++i)
                stage_b128(&VsT[1 - p][vh[i]][vc8],
                           vT + vOff + (size_t)vh[i] * SEQ + kn + vc8);
            stage_wait_n<8>();
        } else {
            stage_wait_n<0>();
        }
#endif
        __syncthreads();

        // S = Q * K^T : 4 key tiles x 4 H-chunks.
        v8f sacc[4];
#pragma unroll
        for (int nt = 0; nt < 4; ++nt) sacc[nt] = (v8f)0.0f;
        const int bsel = (lane >> 4) * 8;
#pragma unroll
        for (int nt = 0; nt < 4; ++nt) {
            int key = nt * 16 + (lane & 15);
#pragma unroll
            for (int c = 0; c < 4; ++c) {
                uint4 blo = *(const uint4*)&Ks[p][key][c * 32 + bsel];
                uint4 bhi = *(const uint4*)&Ks[p][key][c * 32 + bsel + 16];
                sacc[nt] = wmma_bf16(qfrag[c], make_frag(blo, bhi), sacc[nt]);
            }
        }

        // Online softmax: row r lives in vgpr slot r%8 across one 16-lane half.
        float mnew[8], corr[8];
#pragma unroll
        for (int i = 0; i < 8; ++i) {
            float m = fmaxf(fmaxf(sacc[0][i], sacc[1][i]),
                            fmaxf(sacc[2][i], sacc[3][i]));
#pragma unroll
            for (int d = 1; d < 16; d <<= 1) m = fmaxf(m, __shfl_xor(m, d, 32));
            mnew[i] = fmaxf(mrow[i], m);
            corr[i] = __expf(mrow[i] - mnew[i]);
            mrow[i] = mnew[i];
        }
#pragma unroll
        for (int i = 0; i < 8; ++i) {
            float rs = 0.0f;
#pragma unroll
            for (int nt = 0; nt < 4; ++nt) {
                float pv = __expf(sacc[nt][i] - mnew[i]);
                rs += pv;
                Pbuf[wave][i + 8 * (lane >> 4)][nt * 16 + (lane & 15)] = f2bf(pv);
            }
#pragma unroll
            for (int d = 1; d < 16; d <<= 1) rs += __shfl_xor(rs, d, 32);
            lrow[i] = lrow[i] * corr[i] + rs;
#pragma unroll
            for (int t = 0; t < 8; ++t) oacc[t][i] *= corr[i];
        }
        __syncthreads();   // order Pbuf stores before fragment gathers

        // O += P * V : P as 2 A fragments, V as 8x2 B fragments.
        const int asel = (lane >> 4) * 8;
        const int arow = lane & 15;
        v16bf pfrag[2];
#pragma unroll
        for (int c = 0; c < 2; ++c) {
            uint4 lo = *(const uint4*)&Pbuf[wave][arow][c * 32 + asel];
            uint4 hi = *(const uint4*)&Pbuf[wave][arow][c * 32 + asel + 16];
            pfrag[c] = make_frag(lo, hi);
        }
#pragma unroll
        for (int t = 0; t < 8; ++t) {
            int h = t * 16 + (lane & 15);
#pragma unroll
            for (int c = 0; c < 2; ++c) {
                uint4 blo = *(const uint4*)&VsT[p][h][c * 32 + bsel];
                uint4 bhi = *(const uint4*)&VsT[p][h][c * 32 + bsel + 16];
                oacc[t] = wmma_bf16(pfrag[c], make_frag(blo, bhi), oacc[t]);
            }
        }
        __syncthreads();   // all reads of buffer p done before refill
    }

    // Normalize and write fp32 output.
#pragma unroll
    for (int i = 0; i < 8; ++i) {
        float inv = 1.0f / lrow[i];
        int row = qBase + i + 8 * (lane >> 4);
#pragma unroll
        for (int t = 0; t < 8; ++t) {
            out[bOff + (size_t)row * HDIM + t * 16 + (lane & 15)] = oacc[t][i] * inv;
        }
    }
}

// ---------------------------------------------------------------------------
extern "C" void kernel_launch(void* const* d_in, const int* in_sizes, int n_in,
                              void* d_out, int out_size, void* d_ws, size_t ws_size,
                              hipStream_t stream) {
    (void)in_sizes; (void)n_in; (void)out_size; (void)ws_size;
    const float* x  = (const float*)d_in[0];
    const float* Wq = (const float*)d_in[1];
    const float* Wk = (const float*)d_in[2];
    const float* Wv = (const float*)d_in[3];

    const size_t mC = (size_t)BATCH * SEQ * CDIM;
    const size_t mh = (size_t)BATCH * SEQ * HDIM;
    unsigned short* xbf = (unsigned short*)d_ws;    // 64 MB
    unsigned short* WT  = xbf + mC;                 // 1.5 MB
    unsigned short* qws = WT + (size_t)3 * HDIM * CDIM;
    unsigned short* kws = qws + mh;                 // 4 MB each
    unsigned short* vT  = kws + mh;                 // ~77.5 MB total

    cvt_x_kernel<<<dim3(mC / (256 * 8)), 256, 0, stream>>>(x, xbf);
    cvt_w_kernel<<<dim3((CDIM * HDIM) / 256, 3), 256, 0, stream>>>(Wq, Wk, Wv, WT);

    dim3 g1((BATCH * SEQ) / 128, 3);
    qkv_proj_kernel<<<g1, 256, 0, stream>>>(xbf, WT, qws, kws, vT);

    dim3 g2(SEQ / 128, BATCH);
    flash_attn_kernel<<<g2, 256, 0, stream>>>(qws, kws, vT, (float*)d_out);
}
